// Net_83030307766564
// MI455X (gfx1250) — matrix operand order repaired and checked by reading, not compiled, not run on hardware
//
#include <hip/hip_runtime.h>

typedef _Float16 v16h __attribute__((ext_vector_type(16)));
typedef float    v8f  __attribute__((ext_vector_type(8)));

#define BETA   0.95f
#define THR    1.0f
#define NSTEPS 25
#define BATCH  262144
#define NIN    9
#define NHID   100
#define NOUT   2
#define WPB    8            // waves per block
#define TILES  (BATCH / 16) // 16384 tiles, one wave each

typedef __attribute__((address_space(3))) float lds_float;

// One wave (32 lanes) processes a 16-row batch tile for all 25 timesteps.
// mem1/cur1 live in registers in the v_wmma_f32_16x16x32_f16 A-operand layout:
//   lane L: m = L&15, element e of block b covers h = 32*b + (e&7) + 8*(L>>4) + 16*(e>>3)
// W2^T lives in registers in the B-operand layout:
//   lane L: n = L&15, element e of block b covers k = 32*b + (e&7) + 8*(L>>4) + 16*(e>>3)
// C/D layout: VGPR r, lane L -> (m = r + 8*(L>>4), n = L&15); only n<2 is live.
// Per-step outputs are staged into a per-(wave,step) LDS slot and pushed to
// memory with GLOBAL_STORE_ASYNC_FROM_LDS_B32 (ASYNCcnt, never waited on:
// 50 issues < 63 counter capacity; s_endpgm's implicit wait-idle drains it).
// NOTE: no min-waves hint — forcing 3 waves/SIMD spilled loop-carried state
// to scratch inside the hot loop (worse than 2 spill-free waves).
__global__ __launch_bounds__(32 * WPB) void snn_lif_wmma(
    const float* __restrict__ x,   // [B, 9]
    const float* __restrict__ W1,  // [100, 9]
    const float* __restrict__ b1,  // [100]
    const float* __restrict__ W2,  // [2, 100]
    const float* __restrict__ b2,  // [2]
    float* __restrict__ out)       // spk2[25,B,2] ++ mem2[25,B,2]
{
  __shared__ float sh[WPB][NSTEPS][64];   // 51.2 KB/block; occupancy is VGPR-bound

  const int lane = threadIdx.x & 31;
  const int w    = threadIdx.x >> 5;
  const int hi   = lane >> 4;      // K-split half (0/1)
  const int l15  = lane & 15;      // m for A/mem1, n for B/C/D
  const int tile = blockIdx.x * WPB + w;
  const int row  = tile * 16 + l15;

  // ---- load this lane's x row (lanes L and L+16 share a row; L0/L2 absorbs it)
  float xr[NIN];
#pragma unroll
  for (int k = 0; k < NIN; ++k) xr[k] = x[(size_t)row * NIN + k];

  // ---- cur1 = x @ W1^T + b1, computed exactly in f32, directly in A layout
  float cur1v[4][16];
  float mem1v[4][16];
#pragma unroll
  for (int b = 0; b < 4; ++b) {
#pragma unroll
    for (int e = 0; e < 16; ++e) {
      const int kkA = (e & 7) + 16 * (e >> 3);     // compile-time per (b,e)
      if (32 * b + kkA < NHID) {                   // slot live for at least hi=0
        const int h = 32 * b + kkA + 8 * hi;
        float acc = 0.0f;
        if (h < NHID) {
          acc = b1[h];
#pragma unroll
          for (int k = 0; k < NIN; ++k)
            acc = __builtin_fmaf(xr[k], W1[h * NIN + k], acc);
        }
        cur1v[b][e] = acc;
        mem1v[b][e] = 0.0f;
      }
    }
  }

  // ---- W2^T as f16 B operands (only lanes with n<2 carry data; rest zero)
  v16h wB[4];
#pragma unroll
  for (int b = 0; b < 4; ++b) {
#pragma unroll
    for (int e = 0; e < 16; ++e) {
      const int k = 32 * b + (e & 7) + 8 * hi + 16 * (e >> 3);
      float wv = 0.0f;
      if (l15 < NOUT && k < NHID) wv = W2[l15 * NHID + k];
      wB[b][e] = (_Float16)wv;
    }
  }

  // ---- b2 broadcast for C init (zero on dead columns)
  float b2n = 0.0f;
  if (l15 < NOUT) b2n = b2[l15];

  float mem2v[8];
#pragma unroll
  for (int r = 0; r < 8; ++r) mem2v[r] = 0.0f;

  const size_t step_stride = (size_t)BATCH * NOUT;           // 524288
  const size_t mem_half    = (size_t)NSTEPS * step_stride;   // 13107200

#pragma unroll 1
  for (int t = 0; t < NSTEPS; ++t) {
    // C <- b2 broadcast
    v8f c;
#pragma unroll
    for (int r = 0; r < 8; ++r) c[r] = b2n;

    // LIF update of hidden layer; pack spk1 into A operands; 4 chained WMMAs.
#pragma unroll
    for (int b = 0; b < 4; ++b) {
      v16h a;
#pragma unroll
      for (int e = 0; e < 16; ++e) {
        const int kkA = (e & 7) + 16 * (e >> 3);
        if (32 * b + kkA < NHID) {
          const float old  = mem1v[b][e];
          const float rst  = (old > THR) ? THR : 0.0f;   // detached reset (old mem)
          const float mnew = __builtin_fmaf(BETA, old, cur1v[b][e]) - rst;
          mem1v[b][e] = mnew;
          a[e] = (mnew > THR) ? (_Float16)1.0f : (_Float16)0.0f;
        } else {
          a[e] = (_Float16)0.0f;                         // K padding 100..127
        }
      }
      // D = A*B + C  (f32 accumulate; 16-bit WMMA co-executes with VALU above)
      c = __builtin_amdgcn_wmma_f32_16x16x32_f16(
              false, a, false, wB[b], (short)0, c, false, false);
    }

    // LIF update of output layer (valid on lanes with n<2; others compute 0s)
    float spkv[8];
#pragma unroll
    for (int r = 0; r < 8; ++r) {
      const float old  = mem2v[r];
      const float rst  = (old > THR) ? THR : 0.0f;
      const float mnew = __builtin_fmaf(BETA, old, c[r]) - rst;
      mem2v[r] = mnew;
      spkv[r]  = (mnew > THR) ? 1.0f : 0.0f;
    }

    // Stage (m,n) -> contiguous [2*m+n] in this step's private LDS slot.
    float* slot = &sh[w][t][0];
    if (l15 < NOUT) {
#pragma unroll
      for (int r = 0; r < 8; ++r) {
        const int m = r + 8 * hi;
        slot[2 * m + l15]      = spkv[r];
        slot[32 + 2 * m + l15] = mem2v[r];
      }
    }

    // Make the 4-lane LDS writes visible to the async engine, then push both
    // 128B blocks straight from LDS to HBM (no VGPR roundtrip, ASYNCcnt).
    asm volatile("s_wait_dscnt 0x0" ::: "memory");

    const size_t base = (size_t)t * step_stride + (size_t)tile * 32 + lane;
    unsigned long long ga_spk = (unsigned long long)(uintptr_t)(out + base);
    unsigned long long ga_mem = (unsigned long long)(uintptr_t)(out + mem_half + base);
    unsigned int lds_spk = (unsigned int)(uintptr_t)(lds_float*)&slot[lane];
    unsigned int lds_mem = (unsigned int)(uintptr_t)(lds_float*)&slot[32 + lane];
    asm volatile("global_store_async_from_lds_b32 %0, %1, off\n\t"
                 "global_store_async_from_lds_b32 %2, %3, off"
                 :
                 : "v"(ga_spk), "v"(lds_spk), "v"(ga_mem), "v"(lds_mem)
                 : "memory");
  }
}

extern "C" void kernel_launch(void* const* d_in, const int* in_sizes, int n_in,
                              void* d_out, int out_size, void* d_ws, size_t ws_size,
                              hipStream_t stream) {
  (void)in_sizes; (void)n_in; (void)out_size; (void)d_ws; (void)ws_size;
  const float* x  = (const float*)d_in[0];
  const float* W1 = (const float*)d_in[1];
  const float* b1 = (const float*)d_in[2];
  const float* W2 = (const float*)d_in[3];
  const float* b2 = (const float*)d_in[4];
  float* out = (float*)d_out;

  dim3 block(32 * WPB);          // 8 waves
  dim3 grid(TILES / WPB);        // 2048 blocks, 16384 tiles total
  snn_lif_wmma<<<grid, block, 0, stream>>>(x, W1, b1, W2, b2, out);
}
